// Seq2Seq_53274774340234
// MI455X (gfx1250) — matrix operand order repaired and checked by reading, not compiled, not run on hardware
//
#include <hip/hip_runtime.h>

// ----------------------------------------------------------------------------
// Seq2Seq LSTM (S=1024 enc steps, T=1024 dec steps, H=2048, V=256) for MI455X.
//
// Strategy:
//  * Precompute P[v][4H] = Wih @ emb[v] + bih + bhh for all 256 vocab entries
//    via a bf16 WMMA GEMM (8192x2048x256) -- removes Wih@x from the recurrence.
//  * Convert Whh (enc+dec) to bf16: 2 x 33.5 MB -> both L2-resident (192 MB L2),
//    so the sequential 2048-step matvec chain streams from L2, not HBM.
//  * Recurrent step kernel: v_wmma_f32_16x16x32_bf16 with h broadcast into all
//    16 B-columns; each 16x16x32 WMMA eats a 1KB Whh tile. Block = 16 hidden
//    units, computes all 4 gate rows + fused LSTM cell update.
//  * Decoder adds proj matvec (8 blocks) + softmax/argmax (1 block) that writes
//    greedy token feedback into workspace for the next step's kernel.
// Workspace use: ~120 MB.
// ----------------------------------------------------------------------------

constexpr int Hh = 2048;   // hidden
constexpr int H4 = 8192;   // 4*H
constexpr int VV = 256;    // vocab

typedef __attribute__((ext_vector_type(16))) __bf16 v16bf;
typedef __attribute__((ext_vector_type(8)))  float  v8f;

union BFVec { v16bf v; float4 q[2]; };   // 32B: 16 bf16 as two b128 loads
union BF8   { float4 q; __bf16 b[8]; };  // 16B: 8 bf16
union F8    { float4 q[2]; float f[8]; };

__device__ __forceinline__ float sigmoidf_(float x) {
    return 1.0f / (1.0f + __expf(-x));
}

// ---------------------------------------------------------------- conversions
__global__ __launch_bounds__(256)
void f32_to_bf16_x4(const float* __restrict__ s, __bf16* __restrict__ d, int n) {
    int i = (blockIdx.x * 256 + threadIdx.x) * 4;
    if (i + 3 < n) {
        float4 v = *(const float4*)(s + i);
        d[i + 0] = (__bf16)v.x;
        d[i + 1] = (__bf16)v.y;
        d[i + 2] = (__bf16)v.z;
        d[i + 3] = (__bf16)v.w;
    }
}

__global__ __launch_bounds__(256)
void init_state(float* __restrict__ h, float* __restrict__ c,
                __bf16* __restrict__ hbf, int* __restrict__ tok) {
    int i = blockIdx.x * 256 + threadIdx.x;
    if (i < Hh) { h[i] = 0.0f; c[i] = 0.0f; hbf[i] = (__bf16)0.0f; }
    if (i == 0) tok[0] = 0;
}

// --------------------------------------------------- P = Wih @ emb^T + biases
// A: Wih bf16 [H4][Hh] row-major.  B: emb bf16 [VV][Hh] row-major (= B col-major,
// exactly the 32x16 bf16 B-operand layout: lane half h holds k=8h..+7 and 16+8h..+7).
// Out: P [VV][H4] fp32.   One 16x16 tile per wave, K-loop of 64 WMMAs.
__global__ __launch_bounds__(256)
void gemm_P(const __bf16* __restrict__ A, const __bf16* __restrict__ B,
            const float* __restrict__ bih, const float* __restrict__ bhh,
            float* __restrict__ P) {
    const int wave  = blockIdx.x * 8 + (threadIdx.x >> 5);
    const int tileN = wave & 15;        // VV/16 = 16
    const int tileM = wave >> 4;        // H4/16 = 512
    const int lane  = threadIdx.x & 31;
    const int l16   = lane & 15;
    const int kh    = lane >> 4;

    const __bf16* Arow = A + (size_t)(tileM * 16 + l16) * Hh;
    const __bf16* Brow = B + (size_t)(tileN * 16 + l16) * Hh;

    v8f acc = {};
    for (int k0 = 0; k0 < Hh; k0 += 32) {
        BFVec a, b;
        a.q[0] = *(const float4*)(Arow + k0 + kh * 8);        // A: K=kh*8..+7
        a.q[1] = *(const float4*)(Arow + k0 + 16 + kh * 8);   // A: K=16+kh*8..+7
        b.q[0] = *(const float4*)(Brow + k0 + kh * 8);        // B: K=kh*8..+7
        b.q[1] = *(const float4*)(Brow + k0 + 16 + kh * 8);   // B: K=16+kh*8..+7
        acc = __builtin_amdgcn_wmma_f32_16x16x32_bf16(
                false, a.v, false, b.v, (short)0, acc, false, false);
    }
    const int n = tileN * 16 + l16;
#pragma unroll
    for (int j = 0; j < 8; ++j) {
        int m = tileM * 16 + kh * 8 + j;          // D: VGPR j -> M = (lane/16)*8+j
        P[(size_t)n * H4 + m] = acc[j] + bih[m] + bhh[m];
    }
}

// ------------------------------------------------------------- LSTM time step
// gates = P[tok] + Whh @ h; fused cell update. Grid = H/16 = 128 blocks x 256.
// Block owns hidden units j0..j0+15. Wave w: gate g=w&3, K-half kc=w>>2.
// WMMA B operand = h broadcast into all 16 columns -> D replicated over N;
// lanes with (lane&15)==0 carry the 16 row sums of each tile.
__global__ __launch_bounds__(256)
void lstm_step_wmma(const __bf16* __restrict__ Whhbf,  // [H4][Hh] bf16
                    const float*  __restrict__ P,      // [VV][H4]
                    const int*    __restrict__ tokSrc, int tokOff,
                    const float*  __restrict__ c_in,
                    const __bf16* __restrict__ hbf_in, // [Hh] bf16
                    float* __restrict__ h_out, float* __restrict__ c_out,
                    __bf16* __restrict__ hbf_out) {
    __shared__ float partial[4][16][2];
    __shared__ int   tok_s;

    const int tid  = threadIdx.x;
    const int lane = tid & 31;
    const int w    = tid >> 5;     // wave 0..7
    const int g    = w & 3;        // gate i/f/g/o
    const int kc   = w >> 2;       // K half: 0 or 1
    const int j0   = blockIdx.x * 16;
    const int ml   = lane & 15;
    const int kh   = lane >> 4;

    if (tid == 0) tok_s = tokSrc[tokOff];

    const __bf16* Arow = Whhbf + (size_t)(g * Hh + j0 + ml) * Hh;
    const int kbeg = kc * (Hh / 2), kend = kbeg + (Hh / 2);

    v8f acc = {};
    for (int k0 = kbeg; k0 < kend; k0 += 32) {
        BFVec a, b;
        a.q[0] = *(const float4*)(Arow + k0 + kh * 8);
        a.q[1] = *(const float4*)(Arow + k0 + 16 + kh * 8);
        const __bf16* hb = hbf_in + k0 + kh * 8;    // broadcast same h to all cols
        b.q[0] = *(const float4*)(hb);
        b.q[1] = *(const float4*)(hb + 16);
        acc = __builtin_amdgcn_wmma_f32_16x16x32_bf16(
                false, a.v, false, b.v, (short)0, acc, false, false);
    }
    if (ml == 0) {                 // lanes 0 (rows 0..7) and 16 (rows 8..15)
#pragma unroll
        for (int j = 0; j < 8; ++j) partial[g][kh * 8 + j][kc] = acc[j];
    }
    __syncthreads();

    if (tid < 16) {
        const int j = j0 + tid;
        const float* Pr = P + (size_t)tok_s * H4;
        float gi = partial[0][tid][0] + partial[0][tid][1] + Pr[0 * Hh + j];
        float gf = partial[1][tid][0] + partial[1][tid][1] + Pr[1 * Hh + j];
        float gg = partial[2][tid][0] + partial[2][tid][1] + Pr[2 * Hh + j];
        float go = partial[3][tid][0] + partial[3][tid][1] + Pr[3 * Hh + j];
        float iv = sigmoidf_(gi);
        float fv = sigmoidf_(gf);
        float gv = tanhf(gg);
        float ov = sigmoidf_(go);
        float cv = fv * c_in[j] + iv * gv;
        float hv = ov * tanhf(cv);
        c_out[j]   = cv;
        h_out[j]   = hv;
        hbf_out[j] = (__bf16)hv;
    }
}

// ----------------------------------------------------- decoder: logits = Wp@h
__global__ __launch_bounds__(256)
void proj_logits(const __bf16* __restrict__ Wp,   // [VV][Hh] bf16
                 const float* __restrict__ pb,
                 const float* __restrict__ h,
                 float* __restrict__ logits) {
    const int w    = threadIdx.x >> 5;
    const int lane = threadIdx.x & 31;
    const int vbase = blockIdx.x * 32 + w * 4;    // 8 blocks x 8 waves x 4 rows
#pragma unroll
    for (int r = 0; r < 4; ++r) {
        const int v = vbase + r;
        const __bf16* row = Wp + (size_t)v * Hh;
        float acc = 0.0f;
        for (int k0 = lane * 8; k0 < Hh; k0 += 256) {
            BF8 wv; wv.q = *(const float4*)(row + k0);
            F8  hv; hv.q[0] = *(const float4*)(h + k0);
                    hv.q[1] = *(const float4*)(h + k0 + 4);
#pragma unroll
            for (int e = 0; e < 8; ++e) acc += (float)wv.b[e] * hv.f[e];
        }
#pragma unroll
        for (int off = 16; off > 0; off >>= 1) acc += __shfl_xor(acc, off, 32);
        if (lane == 0) logits[v] = acc + pb[v];
    }
}

// -------------------------------------- softmax row -> out, greedy token -> ws
__global__ __launch_bounds__(256)
void softmax_argmax(const float* __restrict__ logits,
                    float* __restrict__ out_row, int* __restrict__ tok) {
    __shared__ float red[256];
    __shared__ int   ridx[256];
    const int t = threadIdx.x;
    const float lg = logits[t];

    red[t] = lg; __syncthreads();
    for (int s = 128; s > 0; s >>= 1) {
        if (t < s) red[t] = fmaxf(red[t], red[t + s]);
        __syncthreads();
    }
    const float mx = red[0]; __syncthreads();

    const float e = __expf(lg - mx);
    red[t] = e; __syncthreads();
    for (int s = 128; s > 0; s >>= 1) {
        if (t < s) red[t] += red[t + s];
        __syncthreads();
    }
    const float sum = red[0]; __syncthreads();
    out_row[t] = e / sum;

    red[t] = lg; ridx[t] = t; __syncthreads();   // argmax, first index on tie
    for (int s = 128; s > 0; s >>= 1) {
        if (t < s) {
            float a = red[t], b = red[t + s];
            if (b > a || (b == a && ridx[t + s] < ridx[t])) {
                red[t] = b; ridx[t] = ridx[t + s];
            }
        }
        __syncthreads();
    }
    if (t == 0) tok[0] = ridx[0];
}

// ---------------------------------------------------------------------- host
extern "C" void kernel_launch(void* const* d_in, const int* in_sizes, int n_in,
                              void* d_out, int out_size, void* d_ws, size_t ws_size,
                              hipStream_t stream) {
    const int*   inputs  = (const int*)  d_in[0];
    const int    S       = in_sizes[0];
    const int    T       = in_sizes[1];
    const float* emb     = (const float*)d_in[2];
    const float* enc_Wih = (const float*)d_in[3];
    const float* enc_Whh = (const float*)d_in[4];
    const float* enc_bih = (const float*)d_in[5];
    const float* enc_bhh = (const float*)d_in[6];
    const float* dec_Wih = (const float*)d_in[7];
    const float* dec_Whh = (const float*)d_in[8];
    const float* dec_bih = (const float*)d_in[9];
    const float* dec_bhh = (const float*)d_in[10];
    const float* proj_W  = (const float*)d_in[11];
    const float* proj_b  = (const float*)d_in[12];
    float* out = (float*)d_out;

    char* ws = (char*)d_ws;
    size_t off = 0;
    auto alloc = [&](size_t bytes) -> void* {
        void* p = ws + off;
        off += (bytes + 255) & ~(size_t)255;
        return p;
    };
    const size_t WBYTES = (size_t)H4 * Hh * sizeof(__bf16);   // 33.5 MB
    __bf16* WhhE   = (__bf16*)alloc(WBYTES);
    __bf16* WhhD   = (__bf16*)alloc(WBYTES);
    __bf16* Wihbf  = (__bf16*)alloc(WBYTES);                  // reused enc->dec
    __bf16* embbf  = (__bf16*)alloc((size_t)VV * Hh * sizeof(__bf16));
    __bf16* projbf = (__bf16*)alloc((size_t)VV * Hh * sizeof(__bf16));
    float*  Penc   = (float*) alloc((size_t)VV * H4 * sizeof(float));
    float*  Pdec   = (float*) alloc((size_t)VV * H4 * sizeof(float));
    float*  hbuf[2]   = { (float*)alloc(Hh * 4), (float*)alloc(Hh * 4) };
    float*  cbuf[2]   = { (float*)alloc(Hh * 4), (float*)alloc(Hh * 4) };
    __bf16* hbfbuf[2] = { (__bf16*)alloc(Hh * 2), (__bf16*)alloc(Hh * 2) };
    float*  logits = (float*)alloc(VV * 4);
    int*    tok    = (int*)  alloc(256);

    const int NW = (int)(H4 * Hh);              // 16.8M weight elements
    const int cgW = NW / 1024;                  // f32_to_bf16_x4 grid
    const int cgE = (VV * Hh) / 1024;

    // Weight conversions (one-time per call)
    f32_to_bf16_x4<<<cgW, 256, 0, stream>>>(enc_Whh, WhhE,  NW);
    f32_to_bf16_x4<<<cgW, 256, 0, stream>>>(dec_Whh, WhhD,  NW);
    f32_to_bf16_x4<<<cgE, 256, 0, stream>>>(emb,     embbf, VV * Hh);
    f32_to_bf16_x4<<<cgE, 256, 0, stream>>>(proj_W,  projbf, VV * Hh);

    // P tables: (H4/16)*(VV/16) = 8192 tiles, 8 waves/block -> 1024 blocks
    f32_to_bf16_x4<<<cgW, 256, 0, stream>>>(enc_Wih, Wihbf, NW);
    gemm_P<<<1024, 256, 0, stream>>>(Wihbf, embbf, enc_bih, enc_bhh, Penc);
    f32_to_bf16_x4<<<cgW, 256, 0, stream>>>(dec_Wih, Wihbf, NW);
    gemm_P<<<1024, 256, 0, stream>>>(Wihbf, embbf, dec_bih, dec_bhh, Pdec);

    init_state<<<(Hh + 255) / 256, 256, 0, stream>>>(hbuf[0], cbuf[0], hbfbuf[0], tok);

    // ---- encoder: 1024 sequential steps (stream-ordered) ----
    for (int t = 0; t < S; ++t) {
        const int pi = t & 1, po = (t + 1) & 1;
        lstm_step_wmma<<<Hh / 16, 256, 0, stream>>>(
            WhhE, Penc, inputs, t,
            cbuf[pi], hbfbuf[pi],
            hbuf[po], cbuf[po], hbfbuf[po]);
    }

    // ---- decoder: step -> project -> softmax/argmax (token feedback) ----
    for (int s = 0; s < T; ++s) {
        const int pi = s & 1, po = (s + 1) & 1;   // S=1024 even: parity continues
        lstm_step_wmma<<<Hh / 16, 256, 0, stream>>>(
            WhhD, Pdec, tok, 0,
            cbuf[pi], hbfbuf[pi],
            hbuf[po], cbuf[po], hbfbuf[po]);
        proj_logits<<<8, 256, 0, stream>>>(projbf, proj_b, hbuf[po], logits);
        softmax_argmax<<<1, 256, 0, stream>>>(logits, out + (size_t)s * VV, tok);
    }
    (void)n_in; (void)out_size; (void)ws_size;
}
